// Nonlocal_51256139710659
// MI455X (gfx1250) — matrix operand order repaired and checked by reading, not compiled, not run on hardware
//
#include <hip/hip_runtime.h>
#include <hip/hip_bf16.h>

typedef __attribute__((ext_vector_type(16))) __bf16 v16bf;
typedef __attribute__((ext_vector_type(8)))  float  v8f;
typedef __attribute__((ext_vector_type(4)))  int    v4i;

#define B_   4
#define C_   256
#define IC_  128
#define H_   64
#define W_   64
#define HW_  4096
#define D_   1152   /* IC*9 */

#define HAS_ASYNC_LDS __has_builtin(__builtin_amdgcn_global_load_async_to_lds_b128)

#if HAS_ASYNC_LDS
typedef __attribute__((address_space(1))) v4i gas_v4i;  // global AS int4
typedef __attribute__((address_space(3))) v4i las_v4i;  // LDS AS int4
typedef __attribute__((address_space(1))) void gas_void;
typedef __attribute__((address_space(3))) void las_void;
#endif

// 16-byte global->LDS copy: async (ASYNCcnt, no VGPR round-trip) when the
// gfx1250 builtin exists, else regular load/store pair.
__device__ __forceinline__ void cp_b128(void* lds, const void* gp) {
#if HAS_ASYNC_LDS
  __builtin_amdgcn_global_load_async_to_lds_b128(
      (gas_v4i*)(gas_void*)const_cast<void*>(gp),
      (las_v4i*)(las_void*)lds, 0, 0);
#else
  *(uint4*)lds = *(const uint4*)gp;
#endif
}

__device__ __forceinline__ void wait_async0() {
#if HAS_ASYNC_LDS
#if __has_builtin(__builtin_amdgcn_s_wait_asynccnt)
  __builtin_amdgcn_s_wait_asynccnt(0);
#else
  asm volatile("s_wait_asynccnt 0x0" ::: "memory");
#endif
#endif
}

__device__ __forceinline__ int reflect_idx(int i, int n) {
  if (i < 0) i = -i;
  if (i >= n) i = 2 * n - 2 - i;
  return i;
}

// ---------------------------------------------------------------------------
// Instance-norm statistics: mean and 1/sqrt(var_unbiased + eps) per (b,c)
// gridDim.x = 2*B*C  (which==0 -> content, which==1 -> style)
// ---------------------------------------------------------------------------
__global__ __launch_bounds__(256) void stats_kernel(
    const float* __restrict__ content, const float* __restrict__ style,
    float* __restrict__ meanC, float* __restrict__ rstdC,
    float* __restrict__ meanS, float* __restrict__ rstdS) {
  __shared__ float ssum[256];
  __shared__ float ssq[256];
  const int idx = blockIdx.x;
  const int which = idx >> 10;          // B*C = 1024
  const int bc = idx & 1023;
  const float* src = (which ? style : content) + (size_t)bc * HW_;
  const int tid = threadIdx.x;
  float s = 0.f, q = 0.f;
  for (int i = tid; i < HW_; i += 256) {
    float v = src[i];
    s += v;
    q += v * v;
  }
  ssum[tid] = s; ssq[tid] = q;
  __syncthreads();
  for (int st = 128; st > 0; st >>= 1) {
    if (tid < st) { ssum[tid] += ssum[tid + st]; ssq[tid] += ssq[tid + st]; }
    __syncthreads();
  }
  if (tid == 0) {
    float m = ssum[0] * (1.f / 4096.f);
    float var = (ssq[0] - 4096.f * m * m) * (1.f / 4095.f);  // ddof=1
    float rs = rsqrtf(var + 1e-5f);
    if (which) { meanS[bc] = m; rstdS[bc] = rs; }
    else       { meanC[bc] = m; rstdC[bc] = rs; }
  }
}

// ---------------------------------------------------------------------------
// t[b][ic][p] = bias[ic] + sum_ch w[ic][ch] * (x[b][ch][p]-mean)*rstd
// i = ((b*IC+ic)*HW + p); coalesced over p.
// ---------------------------------------------------------------------------
__global__ __launch_bounds__(256) void proj_norm_kernel(
    const float* __restrict__ x, const float* __restrict__ w,
    const float* __restrict__ bias, const float* __restrict__ mean,
    const float* __restrict__ rstd, float* __restrict__ out) {
  const int i = blockIdx.x * 256 + threadIdx.x;
  const int p = i & (HW_ - 1);
  const int ic = (i >> 12) & (IC_ - 1);
  const int b = i >> 19;
  const float* xb = x + (size_t)b * C_ * HW_ + p;
  const float* wr = w + ic * C_;
  const float* mb = mean + b * C_;
  const float* rb = rstd + b * C_;
  float acc = bias[ic];
  for (int ch = 0; ch < C_; ++ch)
    acc += wr[ch] * (xb[(size_t)ch * HW_] - mb[ch]) * rb[ch];
  out[i] = acc;
}

// ---------------------------------------------------------------------------
// V stored TRANSPOSED: V[b][ic][p] = bf16(bias[ic] + sum_ch g_w[ic][ch]*x[b][ch][p])
// i = ((b*IC+ic)*HW + p); coalesced fp32 reads and bf16 writes over p.
// (b,ic,p) layout makes the attention V-tile a contiguous b128 LDS copy.
// ---------------------------------------------------------------------------
__global__ __launch_bounds__(256) void proj_g_kernel(
    const float* __restrict__ x, const float* __restrict__ w,
    const float* __restrict__ bias, __bf16* __restrict__ V) {
  const int i = blockIdx.x * 256 + threadIdx.x;
  const int p = i & (HW_ - 1);
  const int ic = (i >> 12) & (IC_ - 1);
  const int b = i >> 19;
  const float* xb = x + (size_t)b * C_ * HW_ + p;
  const float* wr = w + ic * C_;
  float acc = bias[ic];
  for (int ch = 0; ch < C_; ++ch)
    acc += wr[ch] * xb[(size_t)ch * HW_];
  V[i] = (__bf16)acc;
}

// ---------------------------------------------------------------------------
// im2col 3x3 reflection-pad, channel-major feature order f = c*9 + tap.
// out[b][p][f] = bf16( t[b][c][reflect(p + tap)] )
// ---------------------------------------------------------------------------
__global__ __launch_bounds__(256) void im2col_kernel(
    const float* __restrict__ t, __bf16* __restrict__ out) {
  const int i = blockIdx.x * 256 + threadIdx.x;
  const int f = i % D_;
  const int rest = i / D_;
  const int p = rest & (HW_ - 1);
  const int b = rest >> 12;
  const int c = f / 9;
  const int tap = f - c * 9;
  const int dy = tap / 3 - 1;
  const int dx = tap - (tap / 3) * 3 - 1;
  const int y = p >> 6, x = p & 63;
  const int ny = reflect_idx(y + dy, H_);
  const int nx = reflect_idx(x + dx, W_);
  out[i] = (__bf16)t[((size_t)b * IC_ + c) * HW_ + ny * W_ + nx];
}

// ---------------------------------------------------------------------------
// Fused flash-attention: Y[b][q][ic] = softmax(Q K^T)[q,:] @ V  (fp32 out)
// 256 threads = 8 waves; 128 queries/block; 64-key tiles; D=1152 streamed in
// 32-wide chunks. All LDS staging via GLOBAL_LOAD_ASYNC_TO_LDS_B128.
// WMMA: v_wmma_f32_16x16x32_bf16 for both S = QK^T and O += P V.
// ---------------------------------------------------------------------------
__global__ __launch_bounds__(256) void attn_kernel(
    const __bf16* __restrict__ Q, const __bf16* __restrict__ K,
    const __bf16* __restrict__ V, float* __restrict__ Y) {
  // strides padded to 16B multiples (b128 copies) that stay bank-conflict-free
  __shared__ __align__(16) __bf16 sQ[128][40];   // 32-chunk of Q (80 B rows)
  __shared__ __align__(16) __bf16 sK[64][40];    // 32-chunk of K tile
  __shared__ __align__(16) __bf16 sVt[128][72];  // V tile [ic][key] (144 B rows)
  __shared__ __align__(16) __bf16 sP[8][16][66]; // per-wave P (16x64)

  const int tid = threadIdx.x;
  const int wv = tid >> 5;
  const int lane = tid & 31;
  const int hi = lane >> 4;       // lane half (ISA 16-bit fragment layout)
  const int ln = lane & 15;
  const int b = blockIdx.x >> 5;  // HW/128 = 32 q-blocks per batch
  const int q0 = (blockIdx.x & 31) << 7;

  const char* Qg = (const char*)(Q + (size_t)b * HW_ * D_);
  const char* Kg = (const char*)(K + (size_t)b * HW_ * D_);
  const char* Vg = (const char*)(V + (size_t)b * IC_ * HW_);
  char* sQb = (char*)&sQ[0][0];
  char* sKb = (char*)&sK[0][0];
  char* sVb = (char*)&sVt[0][0];

  const v8f vzero = {0.f, 0.f, 0.f, 0.f, 0.f, 0.f, 0.f, 0.f};
  float m_i[8], l_i[8];
  v8f acc[8];  // 8 ic-blocks of 16 -> IC=128
#pragma unroll
  for (int r = 0; r < 8; ++r) { m_i[r] = -3.0e38f; l_i[r] = 0.f; }
#pragma unroll
  for (int i = 0; i < 8; ++i) acc[i] = vzero;

  for (int kt = 0; kt < HW_ / 64; ++kt) {
    const int k0 = kt * 64;
    __syncthreads();  // previous tile's sVt/sP consumers done
    // async V-tile copy: 128 rows x 64 keys bf16 = 1024 x b128
#pragma unroll
    for (int t = 0; t < 4; ++t) {
      int i = tid + (t << 8);
      int ic = i >> 3, s8 = i & 7;
      cp_b128(sVb + ic * 144 + s8 * 16,
              Vg + ((size_t)ic * HW_ + k0) * 2 + s8 * 16);
    }

    v8f s[4];  // S fragments: 4 key-column blocks of 16
#pragma unroll
    for (int cb = 0; cb < 4; ++cb) s[cb] = vzero;

    for (int kc = 0; kc < D_; kc += 32) {
      __syncthreads();  // all waves done reading previous sQ/sK chunk
      // async Q chunk: 128 rows x 64 B = 512 x b128
#pragma unroll
      for (int t = 0; t < 2; ++t) {
        int i = tid + (t << 8);
        int row = i >> 2, s4 = i & 3;
        cp_b128(sQb + row * 80 + s4 * 16,
                Qg + (size_t)(q0 + row) * (D_ * 2) + kc * 2 + s4 * 16);
      }
      // async K chunk: 64 rows x 64 B = 256 x b128
      {
        int row = tid >> 2, s4 = tid & 3;
        cp_b128(sKb + row * 80 + s4 * 16,
                Kg + (size_t)(k0 + row) * (D_ * 2) + kc * 2 + s4 * 16);
      }
      wait_async0();
      __syncthreads();

      // A fragment (Q rows of this wave), ISA 16-bit 16x32 A layout
      union { v16bf v; unsigned int u[8]; } fa;
      const unsigned int* qrow = (const unsigned int*)&sQ[wv * 16 + ln][0];
      const int kbd = hi << 2;  // +8 halves for upper lane half
#pragma unroll
      for (int j = 0; j < 4; ++j) fa.u[j] = qrow[kbd + j];         // K 0..7 / 8..15
#pragma unroll
      for (int j = 0; j < 4; ++j) fa.u[4 + j] = qrow[8 + kbd + j]; // K 16..23 / 24..31

#pragma unroll
      for (int cb = 0; cb < 4; ++cb) {
        // B fragment (K^T), same K order as A: lanes 0-15 K=0..7,16..23; 16-31 K=8..15,24..31
        union { v16bf v; unsigned int u[8]; } fb;
        const unsigned int* krow = (const unsigned int*)&sK[cb * 16 + ln][0];
        const int koff = hi << 2;
#pragma unroll
        for (int j = 0; j < 4; ++j) { fb.u[j] = krow[koff + j]; fb.u[4 + j] = krow[8 + koff + j]; }
        s[cb] = __builtin_amdgcn_wmma_f32_16x16x32_bf16(
            false, fa.v, false, fb.v, (short)0, s[cb], false, false);
      }
    }

    // ---- online softmax over this 64-key tile (rows = r + 8*hi) ----
    float mx[8];
#pragma unroll
    for (int r = 0; r < 8; ++r)
      mx[r] = fmaxf(fmaxf(s[0][r], s[1][r]), fmaxf(s[2][r], s[3][r]));
#pragma unroll
    for (int d = 1; d <= 8; d <<= 1)
#pragma unroll
      for (int r = 0; r < 8; ++r)
        mx[r] = fmaxf(mx[r], __shfl_xor(mx[r], d, 32));

    float rs[8];
#pragma unroll
    for (int r = 0; r < 8; ++r) {
      float mn = fmaxf(m_i[r], mx[r]);
      float sc = __expf(m_i[r] - mn);
      m_i[r] = mn;
      l_i[r] *= sc;
#pragma unroll
      for (int icb = 0; icb < 8; ++icb) acc[icb][r] *= sc;
      rs[r] = 0.f;
    }
#pragma unroll
    for (int cb = 0; cb < 4; ++cb)
#pragma unroll
      for (int r = 0; r < 8; ++r) {
        float pe = __expf(s[cb][r] - m_i[r]);
        s[cb][r] = pe;
        rs[r] += pe;
      }
#pragma unroll
    for (int d = 1; d <= 8; d <<= 1)
#pragma unroll
      for (int r = 0; r < 8; ++r)
        rs[r] += __shfl_xor(rs[r], d, 32);
#pragma unroll
    for (int r = 0; r < 8; ++r) l_i[r] += rs[r];

    // spill P (bf16) to per-wave LDS region for A-layout reload
#pragma unroll
    for (int cb = 0; cb < 4; ++cb)
#pragma unroll
      for (int r = 0; r < 8; ++r)
        sP[wv][r + (hi << 3)][cb * 16 + ln] = (__bf16)s[cb][r];
    __syncthreads();

    // ---- O += P (16x64) x V (64x128) ----
#pragma unroll
    for (int kk = 0; kk < 2; ++kk) {
      union { v16bf v; unsigned int u[8]; } fa2;
      const unsigned int* prow = (const unsigned int*)&sP[wv][ln][0];
      const int base = kk << 4;  // dwords: kk*32 keys
      const int kbd = hi << 2;
#pragma unroll
      for (int j = 0; j < 4; ++j) fa2.u[j] = prow[base + kbd + j];
#pragma unroll
      for (int j = 0; j < 4; ++j) fa2.u[4 + j] = prow[base + 8 + kbd + j];
#pragma unroll
      for (int icb = 0; icb < 8; ++icb) {
        union { v16bf v; unsigned int u[8]; } fb2;
        const unsigned int* vrow = (const unsigned int*)&sVt[icb * 16 + ln][0];
        const int koff = (kk << 4) + (hi << 2);
#pragma unroll
        for (int j = 0; j < 4; ++j) { fb2.u[j] = vrow[koff + j]; fb2.u[4 + j] = vrow[8 + koff + j]; }
        acc[icb] = __builtin_amdgcn_wmma_f32_16x16x32_bf16(
            false, fa2.v, false, fb2.v, (short)0, acc[icb], false, false);
      }
    }
  }

  // ---- epilogue: Y = acc / l ----
#pragma unroll
  for (int icb = 0; icb < 8; ++icb)
#pragma unroll
    for (int r = 0; r < 8; ++r) {
      int q = q0 + wv * 16 + r + (hi << 3);
      int ic = icb * 16 + ln;
      Y[((size_t)b * HW_ + q) * IC_ + ic] = acc[icb][r] / l_i[r];
    }
}

// ---------------------------------------------------------------------------
// out[b][c][p] = W_b[c] + sum_ic W_w[c][ic] * Y[b][p][ic]
// ---------------------------------------------------------------------------
__global__ __launch_bounds__(256) void wout_kernel(
    const float* __restrict__ Y, const float* __restrict__ Ww,
    const float* __restrict__ Wb, float* __restrict__ out) {
  const int i = blockIdx.x * 256 + threadIdx.x;
  const int p = i & (HW_ - 1);
  const int c = (i >> 12) & (C_ - 1);
  const int b = i >> 20;
  const float* yrow = Y + ((size_t)b * HW_ + p) * IC_;
  const float* wr = Ww + c * IC_;
  float acc = Wb[c];
  for (int ic = 0; ic < IC_; ++ic) acc += wr[ic] * yrow[ic];
  out[i] = acc;
}

// ---------------------------------------------------------------------------
extern "C" void kernel_launch(void* const* d_in, const int* in_sizes, int n_in,
                              void* d_out, int out_size, void* d_ws, size_t ws_size,
                              hipStream_t stream) {
  (void)in_sizes; (void)n_in; (void)out_size; (void)ws_size;
  const float* content = (const float*)d_in[0];
  const float* style   = (const float*)d_in[1];
  const float* fusion  = (const float*)d_in[2];
  const float* theta_w = (const float*)d_in[3];
  const float* theta_b = (const float*)d_in[4];
  const float* phi_w   = (const float*)d_in[5];
  const float* phi_b   = (const float*)d_in[6];
  const float* g_w     = (const float*)d_in[7];
  const float* g_b     = (const float*)d_in[8];
  const float* Ww      = (const float*)d_in[9];
  const float* Wb      = (const float*)d_in[10];
  float* out = (float*)d_out;

  char* p = (char*)d_ws;
  float* meanC = (float*)p;  p += (size_t)B_ * C_ * 4;
  float* rstdC = (float*)p;  p += (size_t)B_ * C_ * 4;
  float* meanS = (float*)p;  p += (size_t)B_ * C_ * 4;
  float* rstdS = (float*)p;  p += (size_t)B_ * C_ * 4;
  float* tTheta = (float*)p; p += (size_t)B_ * IC_ * HW_ * 4;
  float* tPhi   = (float*)p; p += (size_t)B_ * IC_ * HW_ * 4;
  float* Ybuf   = (float*)p; p += (size_t)B_ * HW_ * IC_ * 4;
  __bf16* Qbuf = (__bf16*)p; p += (size_t)B_ * HW_ * D_ * 2;
  __bf16* Kbuf = (__bf16*)p; p += (size_t)B_ * HW_ * D_ * 2;
  __bf16* Vbuf = (__bf16*)p; p += (size_t)B_ * IC_ * HW_ * 2;

  stats_kernel<<<2 * B_ * C_, 256, 0, stream>>>(content, style, meanC, rstdC,
                                                meanS, rstdS);
  proj_norm_kernel<<<(B_ * IC_ * HW_) / 256, 256, 0, stream>>>(
      content, theta_w, theta_b, meanC, rstdC, tTheta);
  proj_norm_kernel<<<(B_ * IC_ * HW_) / 256, 256, 0, stream>>>(
      style, phi_w, phi_b, meanS, rstdS, tPhi);
  proj_g_kernel<<<(B_ * IC_ * HW_) / 256, 256, 0, stream>>>(fusion, g_w, g_b,
                                                            Vbuf);
  im2col_kernel<<<(B_ * HW_ * D_) / 256, 256, 0, stream>>>(tTheta, Qbuf);
  im2col_kernel<<<(B_ * HW_ * D_) / 256, 256, 0, stream>>>(tPhi, Kbuf);
  attn_kernel<<<B_ * (HW_ / 128), 256, 0, stream>>>(Qbuf, Kbuf, Vbuf, Ybuf);
  wout_kernel<<<(B_ * C_ * HW_) / 256, 256, 0, stream>>>(Ybuf, Ww, Wb, out);
}
